// ACModel_84808424227462
// MI455X (gfx1250) — matrix operand order repaired and checked by reading, not compiled, not run on hardware
//
#include <hip/hip_runtime.h>

// MI455X / gfx1250, wave32. Forward-value simplification: softmax+one-hot
// collapses to argmax of (e_c . normalized_codebook); e_c normalization is a
// positive per-row scale and cannot change the argmax, so it is skipped.

#define B_SZ   8
#define T_SZ   2048
#define DIN    1024
#define DCODE  256
#define KCB    8192
#define BT     (B_SZ * T_SZ)

typedef __attribute__((ext_vector_type(16))) __bf16 v16bf;
typedef __attribute__((ext_vector_type(8)))  float  v8f;

struct __align__(16) U4 { unsigned x, y, z, w; };
struct __align__(16) F4 { float x, y, z, w; };
union  FragU { v16bf v; U4 q[2]; };

__device__ __forceinline__ unsigned short f2bfb(float f) {
  union { float f; unsigned u; } c; c.f = f;
  unsigned u = c.u;
  unsigned r = u + 0x7FFFu + ((u >> 16) & 1u);   // round-to-nearest-even
  return (unsigned short)(r >> 16);
}

// A-matrix 16x32 bf16 fragment (ISA 7.12.2): lane half h: K chunks {k0+8h..+7} and {k0+16+8h..+7}
__device__ __forceinline__ v16bf load_fragA(const unsigned short* row, int k0, int half) {
  FragU f;
  f.q[0] = *(const U4*)(row + k0 + half * 8);
  f.q[1] = *(const U4*)(row + k0 + 16 + half * 8);
  return f.v;
}
// B-matrix 32x16 bf16 fragment: lane half h: K chunks {k0+8h..+7} and {k0+16+8h..+7}
__device__ __forceinline__ v16bf load_fragB(const unsigned short* row, int k0, int half) {
  FragU f;
  f.q[0] = *(const U4*)(row + k0 + half * 8);
  f.q[1] = *(const U4*)(row + k0 + 16 + half * 8);
  return f.v;
}

__device__ __forceinline__ v8f wmma_bf16(v16bf a, v16bf b, v8f c) {
  return __builtin_amdgcn_wmma_f32_16x16x32_bf16(false, a, false, b, (short)0, c, false, false);
}

// ---------------- prep: fp32 -> bf16 elementwise ----------------
__global__ void cvt_bf16_kernel(const float* __restrict__ src,
                                unsigned short* __restrict__ dst, int n) {
  int i = blockIdx.x * 256 + threadIdx.x;
  if (i < n) dst[i] = f2bfb(src[i]);
}

// ---------------- prep: codebook L2-normalize (+ raw) to bf16 ----------------
// one wave per row; 8 waves per block
__global__ __launch_bounds__(256) void cbnorm_kernel(const float* __restrict__ cb,
                                                     unsigned short* __restrict__ cbn,
                                                     unsigned short* __restrict__ cbraw) {
  int wave = threadIdx.x >> 5, lane = threadIdx.x & 31;
  int row = blockIdx.x * 8 + wave;
  const float* r = cb + (size_t)row * DCODE;
  float vals[8]; float ss = 0.f;
  for (int i = 0; i < 8; ++i) { vals[i] = r[lane + 32 * i]; ss += vals[i] * vals[i]; }
  for (int m = 16; m >= 1; m >>= 1) ss += __shfl_xor(ss, m, 32);
  float inv = 1.0f / fmaxf(sqrtf(ss), 1e-12f);
  for (int i = 0; i < 8; ++i) {
    int c = lane + 32 * i;
    cbn  [(size_t)row * DCODE + c] = f2bfb(vals[i] * inv);
    cbraw[(size_t)row * DCODE + c] = f2bfb(vals[i]);
  }
}

// ---------------- K1: e_c = x @ W_in^T + b_in (bf16 out) ----------------
// block = 16 tokens, 256 threads (8 waves); LDS-staged transposed x tile
#define XS_STRIDE 1032   // 16B-aligned row stride, 4-bank rotation per row
__global__ __launch_bounds__(256) void inproj_kernel(const float* __restrict__ hin,      // [B,DIN,T]
                                                     const unsigned short* __restrict__ Wb, // [DCODE][DIN] bf16
                                                     const float* __restrict__ bias,        // [DCODE]
                                                     unsigned short* __restrict__ ec) {     // [BT][DCODE] bf16
  __shared__ unsigned short xs[16 * XS_STRIDE];
  int blk = blockIdx.x;
  int b  = blk / (T_SZ / 16);
  int t0 = (blk % (T_SZ / 16)) * 16;
  const float* hb = hin + (size_t)b * DIN * T_SZ;

  for (int i = threadIdx.x; i < 16 * DIN; i += 256) {
    int tl = i & 15, d = i >> 4;                       // coalesced along t
    xs[tl * XS_STRIDE + d] = f2bfb(hb[(size_t)d * T_SZ + t0 + tl]);
  }
  __syncthreads();

  int wave = threadIdx.x >> 5, lane = threadIdx.x & 31;
  int h4 = lane >> 4, ln = lane & 15;
  int c0 = wave * 32;
  const unsigned short* a_row  = xs + ln * XS_STRIDE;
  const unsigned short* b_row0 = Wb + (size_t)(c0 + ln) * DIN;
  const unsigned short* b_row1 = Wb + (size_t)(c0 + 16 + ln) * DIN;

  v8f acc0 = {}; v8f acc1 = {};
  for (int k0 = 0; k0 < DIN; k0 += 32) {
    v16bf a  = load_fragA(a_row, k0, h4);
    v16bf b0 = load_fragB(b_row0, k0, h4);
    v16bf b1 = load_fragB(b_row1, k0, h4);
    acc0 = wmma_bf16(a, b0, acc0);
    acc1 = wmma_bf16(a, b1, acc1);
  }
  float bv0 = bias[c0 + ln], bv1 = bias[c0 + 16 + ln];
  int tok0 = b * T_SZ + t0 + h4 * 8;
  for (int r = 0; r < 8; ++r) {
    size_t base = (size_t)(tok0 + r) * DCODE;
    ec[base + c0 + ln]      = f2bfb(acc0[r] + bv0);
    ec[base + c0 + 16 + ln] = f2bfb(acc1[r] + bv1);
  }
}

// ---------------- K2: argmax_k (e_c . cb_norm_k) via WMMA ----------------
// block = 32 tokens; each of 8 waves scans 1024 codes (64 tiles x 16 WMMA).
// Two 16-token A tiles live in registers -> each B fragment feeds TWO WMMAs
// (1 global_load_b128 per wmma instead of 2); B (4 MB bf16) stays L2-resident.
__global__ __launch_bounds__(256) void argmax_kernel(const unsigned short* __restrict__ ec,  // [BT][DCODE]
                                                     const unsigned short* __restrict__ cbn, // [KCB][DCODE]
                                                     int* __restrict__ idx_out) {            // [BT]
  int row0 = blockIdx.x * 32;
  int wave = threadIdx.x >> 5, lane = threadIdx.x & 31;
  int h4 = lane >> 4, ln = lane & 15;

  const unsigned short* a_row0 = ec + (size_t)(row0 + ln) * DCODE;
  const unsigned short* a_row1 = ec + (size_t)(row0 + 16 + ln) * DCODE;
  v16bf a0[8], a1[8];
  for (int kk = 0; kk < 8; ++kk) {
    a0[kk] = load_fragA(a_row0, kk * 32, h4);
    a1[kk] = load_fragA(a_row1, kk * 32, h4);
  }

  float maxv0[8], maxv1[8]; int maxi0[8], maxi1[8];
  for (int r = 0; r < 8; ++r) {
    maxv0[r] = -3.4e38f; maxi0[r] = 0;
    maxv1[r] = -3.4e38f; maxi1[r] = 0;
  }

  int cbase = wave * (KCB / 8);
  const int ntiles = (KCB / 8) / 16;
  for (int ct = 0; ct < ntiles; ++ct) {
    int ct0 = cbase + ct * 16;
    const unsigned short* brow = cbn + (size_t)(ct0 + ln) * DCODE;
    if (ct + 1 < ntiles)
      __builtin_prefetch(brow + 16 * DCODE, 0, 1);     // -> global_prefetch_b8 (next code tile)
    v8f acc0 = {}; v8f acc1 = {};
    for (int kk = 0; kk < 8; ++kk) {
      v16bf bb = load_fragB(brow, kk * 32, h4);
      acc0 = wmma_bf16(a0[kk], bb, acc0);
      acc1 = wmma_bf16(a1[kk], bb, acc1);
    }
    int ci = ct0 + ln;
    for (int r = 0; r < 8; ++r) {
      if (acc0[r] > maxv0[r]) { maxv0[r] = acc0[r]; maxi0[r] = ci; }
      if (acc1[r] > maxv1[r]) { maxv1[r] = acc1[r]; maxi1[r] = ci; }
    }
  }

  // reduce across the 16 columns held by each half-wave (xor masks stay in-half)
  for (int m = 8; m >= 1; m >>= 1) {
    for (int r = 0; r < 8; ++r) {
      float ov = __shfl_xor(maxv0[r], m, 32);
      int   oi = __shfl_xor(maxi0[r], m, 32);
      if (ov > maxv0[r] || (ov == maxv0[r] && oi < maxi0[r])) { maxv0[r] = ov; maxi0[r] = oi; }
      ov = __shfl_xor(maxv1[r], m, 32);
      oi = __shfl_xor(maxi1[r], m, 32);
      if (ov > maxv1[r] || (ov == maxv1[r] && oi < maxi1[r])) { maxv1[r] = ov; maxi1[r] = oi; }
    }
  }

  __shared__ float sval[8][32];
  __shared__ int   sidx[8][32];
  if (ln == 0) {
    for (int r = 0; r < 8; ++r) {
      sval[wave][h4 * 8 + r]      = maxv0[r]; sidx[wave][h4 * 8 + r]      = maxi0[r];
      sval[wave][16 + h4 * 8 + r] = maxv1[r]; sidx[wave][16 + h4 * 8 + r] = maxi1[r];
    }
  }
  __syncthreads();
  if (threadIdx.x < 32) {
    int rr = threadIdx.x;
    float bv = sval[0][rr]; int bi = sidx[0][rr];
    for (int w = 1; w < 8; ++w) {
      float v = sval[w][rr]; int i2 = sidx[w][rr];
      if (v > bv || (v == bv && i2 < bi)) { bv = v; bi = i2; }
    }
    idx_out[row0 + rr] = bi;
  }
}

// ---------------- K3: z = codebook[idx] @ W_out^T + b_out, transposed store ----------------
__global__ __launch_bounds__(256) void outproj_kernel(const int* __restrict__ idx,
                                                      const unsigned short* __restrict__ cbraw, // [KCB][DCODE]
                                                      const unsigned short* __restrict__ Wob,   // [DIN][DCODE]
                                                      const float* __restrict__ bias,           // [DIN]
                                                      float* __restrict__ out) {                // [B,DIN,T]
  int blk = blockIdx.x;
  int b  = blk / (T_SZ / 16);
  int t0 = (blk % (T_SZ / 16)) * 16;
  int row0 = b * T_SZ + t0;
  int wave = threadIdx.x >> 5, lane = threadIdx.x & 31;
  int h4 = lane >> 4, ln = lane & 15;

  int code = idx[row0 + ln];
  const unsigned short* a_row = cbraw + (size_t)code * DCODE;
  v16bf a[8];
  for (int kk = 0; kk < 8; ++kk) a[kk] = load_fragA(a_row, kk * 32, h4);

  int d0 = wave * (DIN / 8);
  for (int j = 0; j < 8; ++j) {
    int dt = d0 + j * 16;
    const unsigned short* brow = Wob + (size_t)(dt + ln) * DCODE;
    v8f acc = {};
    for (int kk = 0; kk < 8; ++kk) {
      v16bf bb = load_fragB(brow, kk * 32, h4);
      acc = wmma_bf16(a[kk], bb, acc);
    }
    int d = dt + ln;
    float bv = bias[d];
    // D-tile slot r -> (token t0 + 8*h4 + r, feature d): 8 consecutive t per lane
    float* p = out + (size_t)b * DIN * T_SZ + (size_t)d * T_SZ + t0 + h4 * 8;
    F4 lo = { acc[0] + bv, acc[1] + bv, acc[2] + bv, acc[3] + bv };
    F4 hi = { acc[4] + bv, acc[5] + bv, acc[6] + bv, acc[7] + bv };
    *(F4*)(p)     = lo;
    *(F4*)(p + 4) = hi;
  }
}

extern "C" void kernel_launch(void* const* d_in, const int* in_sizes, int n_in,
                              void* d_out, int out_size, void* d_ws, size_t ws_size,
                              hipStream_t stream) {
  const float* h_input = (const float*)d_in[0];   // [B,DIN,T]
  const float* in_w    = (const float*)d_in[1];   // [DCODE,DIN]
  const float* in_b    = (const float*)d_in[2];   // [DCODE]
  const float* out_w   = (const float*)d_in[3];   // [DIN,DCODE]
  const float* out_b   = (const float*)d_in[4];   // [DIN]
  const float* cb      = (const float*)d_in[5];   // [KCB,DCODE]

  char* ws = (char*)d_ws;
  size_t off = 0;
  auto carve = [&](size_t bytes) { void* p = ws + off; off = (off + bytes + 255) & ~(size_t)255; return p; };
  unsigned short* Winb  = (unsigned short*)carve((size_t)DCODE * DIN * 2);   // 512 KB
  unsigned short* Wob   = (unsigned short*)carve((size_t)DIN * DCODE * 2);   // 512 KB
  unsigned short* cbn   = (unsigned short*)carve((size_t)KCB * DCODE * 2);   // 4 MB
  unsigned short* cbraw = (unsigned short*)carve((size_t)KCB * DCODE * 2);   // 4 MB
  unsigned short* ec    = (unsigned short*)carve((size_t)BT * DCODE * 2);    // 8 MB
  int*            idx   = (int*)carve((size_t)BT * 4);                       // 64 KB

  int nw = DCODE * DIN;
  cvt_bf16_kernel<<<(nw + 255) / 256, 256, 0, stream>>>(in_w, Winb, nw);
  cvt_bf16_kernel<<<(nw + 255) / 256, 256, 0, stream>>>(out_w, Wob, nw);
  cbnorm_kernel<<<KCB / 8, 256, 0, stream>>>(cb, cbn, cbraw);

  inproj_kernel<<<BT / 16, 256, 0, stream>>>(h_input, Winb, in_b, ec);
  argmax_kernel<<<BT / 32, 256, 0, stream>>>(ec, cbn, idx);
  outproj_kernel<<<BT / 16, 256, 0, stream>>>(idx, cbraw, Wob, out_b, (float*)d_out);
}
